// QLSTM_65481071396036
// MI455X (gfx1250) — matrix operand order repaired and checked by reading, not compiled, not run on hardware
//
#include <hip/hip_runtime.h>

#define SEQ     512
#define BATCH   512
#define IN_DIM  128
#define NQ      8
#define DTOT    136            // IN_DIM + NQ
#define GATES   32             // 4 * NQ
#define MTOT    (SEQ * BATCH)  // 262144 rows of the big GEMM

// ws layout: [0, 32 MiB) = Z (MTOT x 32 f32); [32 MiB, +16 KiB) = B fragments
#define ZBYTES  ((size_t)MTOT * GATES * 4)
// B frag table: [nt(2)][k(4)][hi/lo(2)][lane(32)][16 x bf16] = 16384 bytes

typedef __attribute__((ext_vector_type(16))) __bf16 v16bf;
typedef __attribute__((ext_vector_type(8)))  float  v8f;
typedef __attribute__((ext_vector_type(4)))  float  v4f;   // clang vector: OK for nontemporal builtins

// =======================================================================
// Kernel 0 (prep): pre-swizzle W[:, :128] into WMMA B-operand bf16 hi/lo
// fragments. One thread per (nt, k, lane) triple -> 256 threads total.
//   B lane layout: lane<16 -> N=lane,   K = k0 + 0..15
//                  lane>=16 -> N=lane-16, K = k0 + 16..31
// =======================================================================
__global__ void __launch_bounds__(256)
qlstm_prep_bfrag(const float* __restrict__ W, v16bf* __restrict__ Bf)
{
    const int tid  = threadIdx.x;          // 0..255
    const int lane = tid & 31;
    const int k    = (tid >> 5) & 3;       // k0 = k*32
    const int nt   = tid >> 7;             // 0..1
    const int row  = nt * 16 + (lane & 15);
    const int kb   = k * 32 + (lane >> 4) * 16;

    v16bf hi, lo;
#pragma unroll
    for (int j = 0; j < 16; ++j) {
        float w = W[(size_t)row * DTOT + kb + j];
        __bf16 h = (__bf16)w;              // RNE, native cvt
        hi[j] = h;
        lo[j] = (__bf16)(w - (float)h);
    }
    Bf[(size_t)(((nt * 4 + k) * 2 + 0) * 32 + lane)] = hi;
    Bf[(size_t)(((nt * 4 + k) * 2 + 1) * 32 + lane)] = lo;
}

// =======================================================================
// Kernel 1: Z[m, gq] = sum_{d<128} X[m, d] * W[gq, d]   via split-bf16 WMMA
//   Each wave: 16(M) x 32(N) tile, K loop 0..127 step 32 -> 24 WMMAs.
//   A converted on the fly with native cvt; B loaded pre-swizzled from ws.
// =======================================================================
__global__ void __launch_bounds__(256)
qlstm_gemm_wmma(const float* __restrict__ X,
                const v16bf* __restrict__ Bf,
                float* __restrict__ Z)
{
    const int lane  = threadIdx.x & 31;
    const int wave  = threadIdx.x >> 5;
    const int mtile = blockIdx.x * 8 + wave;     // 16384 tiles total
    const int m0    = mtile * 16;
    const int r     = lane & 15;
    const int half  = lane >> 4;                 // 0 or 1

    v8f acc0 = {};   // N columns 0..15
    v8f acc1 = {};   // N columns 16..31

    // A k order used here (contiguous, matches B): lane<16 -> K = k0 + 0..15
    //                                 lane>=16 -> K = k0 + 16..31
    const float* arow = X + (size_t)(m0 + r) * IN_DIM + half * 16;

#pragma unroll
    for (int kk = 0; kk < 4; ++kk) {
        const int k0 = kk * 32;
        // ---- A fragment: 4 x b128 nontemporal (stream-once, keep Z in L2)
        const v4f* a0 = (const v4f*)(arow + k0);
        const v4f* a1 = (const v4f*)(arow + k0 + 8);
        v4f x0 = __builtin_nontemporal_load(a0);
        v4f x1 = __builtin_nontemporal_load(a0 + 1);
        v4f x2 = __builtin_nontemporal_load(a1);
        v4f x3 = __builtin_nontemporal_load(a1 + 1);
        float xa[16] = { x0[0], x0[1], x0[2], x0[3],  x1[0], x1[1], x1[2], x1[3],
                         x2[0], x2[1], x2[2], x2[3],  x3[0], x3[1], x3[2], x3[3] };
        v16bf a_hi, a_lo;
#pragma unroll
        for (int j = 0; j < 16; ++j) {
            __bf16 h = (__bf16)xa[j];       // v_cvt_pk_bf16_f32 (RNE)
            a_hi[j] = h;
            a_lo[j] = (__bf16)(xa[j] - (float)h);
        }

        // ---- B fragments: pre-swizzled, 2 x b128 each, L0-resident
        v16bf b0h = Bf[(size_t)(((0 * 4 + kk) * 2 + 0) * 32 + lane)];
        v16bf b0l = Bf[(size_t)(((0 * 4 + kk) * 2 + 1) * 32 + lane)];
        v16bf b1h = Bf[(size_t)(((1 * 4 + kk) * 2 + 0) * 32 + lane)];
        v16bf b1l = Bf[(size_t)(((1 * 4 + kk) * 2 + 1) * 32 + lane)];

        acc0 = __builtin_amdgcn_wmma_f32_16x16x32_bf16(false, a_lo, false, b0h, (short)0, acc0, false, false);
        acc0 = __builtin_amdgcn_wmma_f32_16x16x32_bf16(false, a_hi, false, b0l, (short)0, acc0, false, false);
        acc0 = __builtin_amdgcn_wmma_f32_16x16x32_bf16(false, a_hi, false, b0h, (short)0, acc0, false, false);
        acc1 = __builtin_amdgcn_wmma_f32_16x16x32_bf16(false, a_lo, false, b1h, (short)0, acc1, false, false);
        acc1 = __builtin_amdgcn_wmma_f32_16x16x32_bf16(false, a_hi, false, b1l, (short)0, acc1, false, false);
        acc1 = __builtin_amdgcn_wmma_f32_16x16x32_bf16(false, a_hi, false, b1h, (short)0, acc1, false, false);
    }

    // C/D layout: lanes 0-15 -> M = i (VGPR i), lanes 16-31 -> M = 8 + i; N = lane&15
    float* zr = Z + (size_t)(m0 + half * 8) * GATES + r;
#pragma unroll
    for (int i = 0; i < 8; ++i) {
        zr[(size_t)i * GATES]      = acc0[i];
        zr[(size_t)i * GATES + 16] = acc1[i];
    }
}

// =======================================================================
// Kernel 2: serial-in-t recurrence, one thread per batch chain.
//   z = Z[t,b,:] + Wh @ h + (b + qp);  q = cumprod(cos(z)) per gate;
//   c = sig(qf)*c + sig(qi)*tanh(qg);  h = sig(qo)*tanh(c)
// =======================================================================
__global__ void __launch_bounds__(128)
qlstm_recurrence(const float* __restrict__ Z,
                 const float* __restrict__ W,
                 const float* __restrict__ bias,
                 const float* __restrict__ qp,
                 const float* __restrict__ h0,
                 const float* __restrict__ c0,
                 float* __restrict__ out)
{
    __shared__ float sWh[GATES][NQ];
    __shared__ float sBQ[GATES];
    const int tid = threadIdx.x;
    for (int i = tid; i < GATES * NQ; i += blockDim.x)
        sWh[i >> 3][i & 7] = W[(size_t)(i >> 3) * DTOT + IN_DIM + (i & 7)];
    for (int i = tid; i < GATES; i += blockDim.x)
        sBQ[i] = bias[i] + qp[i];
    __syncthreads();

    const int b = blockIdx.x * blockDim.x + tid;   // 0..511
    float h[NQ], c[NQ];
#pragma unroll
    for (int q = 0; q < NQ; ++q) { h[q] = h0[b * NQ + q]; c[q] = c0[b * NQ + q]; }

    const size_t zstride = (size_t)BATCH * GATES;
    const size_t ostride = (size_t)BATCH * NQ;
    const float* zp = Z + (size_t)b * GATES;
    float*       op = out + (size_t)b * NQ;

    // software pipeline: preload t=0
    v4f buf[8];
    {
        const v4f* zv = (const v4f*)zp;
#pragma unroll
        for (int i = 0; i < 8; ++i) buf[i] = zv[i];
    }

    for (int t = 0; t < SEQ; ++t) {
        float z[GATES];
#pragma unroll
        for (int i = 0; i < 8; ++i) {
            z[4 * i + 0] = buf[i][0]; z[4 * i + 1] = buf[i][1];
            z[4 * i + 2] = buf[i][2]; z[4 * i + 3] = buf[i][3];
        }
        // issue next timestep's loads before the dependent compute chain
        if (t + 1 < SEQ) {
            zp += zstride;
            __builtin_prefetch(zp + zstride, 0, 0);   // t+2 into caches
            const v4f* zv = (const v4f*)zp;
#pragma unroll
            for (int i = 0; i < 8; ++i) buf[i] = zv[i];
        }

        // z += Wh @ h + (bias + qp)   (LDS reads are wave-uniform -> broadcast)
#pragma unroll
        for (int gq = 0; gq < GATES; ++gq) {
            float s = z[gq] + sBQ[gq];
#pragma unroll
            for (int j = 0; j < NQ; ++j) s = fmaf(sWh[gq][j], h[j], s);
            z[gq] = s;
        }

        // per-gate running cumprod of cos, fused with cell update
        float pf = 1.f, pi = 1.f, pg = 1.f, po = 1.f;
#pragma unroll
        for (int q = 0; q < NQ; ++q) {
            pf *= __cosf(z[0 * NQ + q]);
            pi *= __cosf(z[1 * NQ + q]);
            pg *= __cosf(z[2 * NQ + q]);
            po *= __cosf(z[3 * NQ + q]);
            float f  = 1.f / (1.f + __expf(-pf));
            float ig = 1.f / (1.f + __expf(-pi));
            float g  = tanhf(pg);
            float o  = 1.f / (1.f + __expf(-po));
            float cn = fmaf(f, c[q], ig * g);
            c[q] = cn;
            h[q] = o * tanhf(cn);
        }

        // store h -> outputs[t, b, :]
        v4f* ov = (v4f*)op;
        v4f lo, hi;
        lo[0] = h[0]; lo[1] = h[1]; lo[2] = h[2]; lo[3] = h[3];
        hi[0] = h[4]; hi[1] = h[5]; hi[2] = h[6]; hi[3] = h[7];
        ov[0] = lo; ov[1] = hi;
        op += ostride;
    }

    // final h, c
    float* hf = out + (size_t)SEQ * BATCH * NQ + (size_t)b * NQ;
    float* cf = hf + (size_t)BATCH * NQ;
#pragma unroll
    for (int q = 0; q < NQ; ++q) { hf[q] = h[q]; cf[q] = c[q]; }
}

// =======================================================================
extern "C" void kernel_launch(void* const* d_in, const int* in_sizes, int n_in,
                              void* d_out, int out_size, void* d_ws, size_t ws_size,
                              hipStream_t stream) {
    (void)in_sizes; (void)n_in; (void)out_size; (void)ws_size;
    const float* X    = (const float*)d_in[0];  // (SEQ, BATCH, 128)
    const float* h0   = (const float*)d_in[1];  // (BATCH, 8)
    const float* c0   = (const float*)d_in[2];  // (BATCH, 8)
    const float* W    = (const float*)d_in[3];  // (4, 8, 136) -> (32, 136)
    const float* bias = (const float*)d_in[4];  // (4, 8) -> (32)
    const float* qp   = (const float*)d_in[5];  // (4, 8) -> (32)
    float* out = (float*)d_out;
    float* Z   = (float*)d_ws;                                   // 32 MiB
    v16bf* Bf  = (v16bf*)((char*)d_ws + ZBYTES);                 // +16 KiB

    qlstm_prep_bfrag<<<dim3(1), dim3(256), 0, stream>>>(W, Bf);
    // GEMM: 16384 wave-tiles, 8 waves (256 threads) per block -> 2048 blocks
    qlstm_gemm_wmma<<<dim3(MTOT / 16 / 8), dim3(256), 0, stream>>>(X, Bf, Z);
    // Recurrence: 512 independent chains, 128 threads/block
    qlstm_recurrence<<<dim3(BATCH / 128), dim3(128), 0, stream>>>(Z, W, bias, qp, h0, c0, out);
}
